// GLM47AttentionOptimizer_88725434401134
// MI455X (gfx1250) — matrix operand order repaired and checked by reading, not compiled, not run on hardware
//
#include <hip/hip_runtime.h>
#include <math.h>

typedef __attribute__((ext_vector_type(16))) _Float16 v16h;
typedef __attribute__((ext_vector_type(8)))  _Float16 v8h;
typedef __attribute__((ext_vector_type(4)))  _Float16 v4h;
typedef __attribute__((ext_vector_type(8)))  float    v8f;
typedef __attribute__((ext_vector_type(4)))  int      v4i;

#define S_  2048
#define H_  2048
#define NH_ 16
#define D_  128
#define KK_ 1433   // int(2048 * (1.0 - 0.3))

#define BM 128
#define BN 128
#define BK 64
#define LDSS 72    // halves per LDS row (BK + 8 pad); row stride 144B (16B aligned)

#if defined(__gfx1250__) && __has_builtin(__builtin_amdgcn_global_load_async_to_lds_b128)
#define USE_ASYNC 1
#else
#define USE_ASYNC 0
#endif

#if USE_ASYNC
typedef __attribute__((address_space(1))) v4i v4i_g;   // global int4
typedef __attribute__((address_space(3))) v4i v4i_l;   // LDS int4
#endif

// 16-byte global -> LDS copy (async on gfx1250 when available)
__device__ __forceinline__ void cp16(_Float16* l, const _Float16* g) {
#if USE_ASYNC
  __builtin_amdgcn_global_load_async_to_lds_b128((v4i_g*)g, (v4i_l*)l, 0, 0);
#else
  *(v8h*)l = *(const v8h*)g;
#endif
}

__device__ __forceinline__ void wait_async() {
#if USE_ASYNC
#if __has_builtin(__builtin_amdgcn_s_wait_asynccnt)
  __builtin_amdgcn_s_wait_asynccnt(0);
#else
  asm volatile("s_wait_asynccnt 0x0" ::: "memory");
#endif
#endif
}

// Load a v16h fragment as two 8-half chunks at p[0..7] and p[off2..off2+7]
__device__ __forceinline__ v16h ld_frag(const _Float16* p, int off2) {
  union { v16h v; v8h h[2]; } u;
  u.h[0] = *(const v8h*)(p);
  u.h[1] = *(const v8h*)(p + off2);
  return u.v;
}

// Batched f16 GEMM, always C = scale * A (MxK) * B^T (B is NxK, row-major along k).
// outF32: C stored fp32 (else f16). cT: store C transposed (C[n*ldc+m]).
__global__ __launch_bounds__(256)
void wmma_gemm(const _Float16* __restrict__ A, const _Float16* __restrict__ B,
               void* __restrict__ C, int K,
               int lda, int ldb, int ldc,
               long sA, long sB, long sC, float scale, int outF32, int cT)
{
  __shared__ _Float16 lA[2 * BM * LDSS];
  __shared__ _Float16 lB[2 * BN * LDSS];
  const int tid = threadIdx.x;
  const int m0 = blockIdx.y * BM, n0 = blockIdx.x * BN;
  const _Float16* gA = A + (size_t)blockIdx.z * sA + (size_t)m0 * lda;
  const _Float16* gB = B + (size_t)blockIdx.z * sB + (size_t)n0 * ldb;

  const int lane = tid & 31, w = tid >> 5;
  const int wm = (w & 3) * 32;      // wave M offset within block tile
  const int wn = (w >> 2) * 64;     // wave N offset within block tile
  const int fr = lane & 15;         // fragment row (m or n)
  const int ka = (lane >> 4) * 8;   // A k-offset: chunks [ka,+8) and [ka+16,+8)
  const int kbo = (lane >> 4) * 8;  // B k-offset: chunks [kbo,+8) and [kbo+16,+8)
  const int sr = tid >> 3;          // staging row base
  const int sc8 = (tid & 7) << 3;   // staging chunk col (halves)

  v8f acc[2][4];
  const v8f vz = {0.f,0.f,0.f,0.f,0.f,0.f,0.f,0.f};
#pragma unroll
  for (int i = 0; i < 2; ++i)
#pragma unroll
    for (int j = 0; j < 4; ++j) acc[i][j] = vz;

  // prologue: stage tile 0 into buffer 0
#pragma unroll
  for (int i = 0; i < 4; ++i) {
    int r = sr + 32 * i;
    cp16(&lA[r * LDSS + sc8], gA + (size_t)r * lda + sc8);
    cp16(&lB[r * LDSS + sc8], gB + (size_t)r * ldb + sc8);
  }

  const int nT = K / BK;
  for (int t = 0; t < nT; ++t) {
    wait_async();
    __syncthreads();
    if (t + 1 < nT) {
      _Float16* nA = &lA[((t + 1) & 1) * BM * LDSS];
      _Float16* nB = &lB[((t + 1) & 1) * BN * LDSS];
      int kk = (t + 1) * BK;
#pragma unroll
      for (int i = 0; i < 4; ++i) {
        int r = sr + 32 * i;
        cp16(&nA[r * LDSS + sc8], gA + (size_t)r * lda + kk + sc8);
        cp16(&nB[r * LDSS + sc8], gB + (size_t)r * ldb + kk + sc8);
      }
    }
    const _Float16* sA = &lA[(t & 1) * BM * LDSS];
    const _Float16* sB = &lB[(t & 1) * BN * LDSS];
#pragma unroll
    for (int ks = 0; ks < 2; ++ks) {     // two 16x16x32 k-steps per tile
      const int kb = ks * 32;
      v16h af[2], bf[4];
#pragma unroll
      for (int i = 0; i < 2; ++i)
        af[i] = ld_frag(&sA[(wm + i * 16 + fr) * LDSS + kb + ka], 16);
#pragma unroll
      for (int j = 0; j < 4; ++j)
        bf[j] = ld_frag(&sB[(wn + j * 16 + fr) * LDSS + kb + kbo], 16);
#pragma unroll
      for (int i = 0; i < 2; ++i)
#pragma unroll
        for (int j = 0; j < 4; ++j)
          acc[i][j] = __builtin_amdgcn_wmma_f32_16x16x32_f16(
              false, af[i], false, bf[j], (short)0, acc[i][j], false, false);
    }
  }

  float* Cf = (float*)C + (size_t)blockIdx.z * sC;
  _Float16* Ch = (_Float16*)C + (size_t)blockIdx.z * sC;
  const int mhi = (lane >> 4) * 8;
#pragma unroll
  for (int i = 0; i < 2; ++i)
#pragma unroll
    for (int j = 0; j < 4; ++j)
#pragma unroll
      for (int r = 0; r < 8; ++r) {
        int m = m0 + wm + i * 16 + r + mhi;
        int n = n0 + wn + j * 16 + fr;
        size_t idx = cT ? (size_t)n * ldc + m : (size_t)m * ldc + n;
        float x = acc[i][j][r] * scale;
        if (outF32) Cf[idx] = x;
        else        Ch[idx] = (_Float16)x;
      }
}

__global__ __launch_bounds__(256)
void f32_to_f16(const float* __restrict__ in, _Float16* __restrict__ out, int n) {
  int i = (blockIdx.x * 256 + threadIdx.x) * 4;
  if (i < n) {
    float4 v = *(const float4*)(in + i);
    v4h h = {(_Float16)v.x, (_Float16)v.y, (_Float16)v.z, (_Float16)v.w};
    *(v4h*)(out + i) = h;
  }
}

// tWT[h][e][d] = tanh(pw[h][d][e])  (pre-transposed so pattern GEMM is A*B^T)
__global__ __launch_bounds__(256)
void tanhT_kernel(const float* __restrict__ pw, _Float16* __restrict__ tWT) {
  int t = blockIdx.x * 256 + threadIdx.x;      // [0, NH*D*D)
  int h = t / (D_ * D_);
  int r = t % (D_ * D_);
  int e = r / D_, d = r % D_;
  tWT[t] = (_Float16)tanhf(pw[h * D_ * D_ + d * D_ + e]);
}

// HF rotate_half RoPE, in-place on f16 Q and K ([S,H] layout, heads of 128).
__global__ __launch_bounds__(256)
void rope_kernel(_Float16* __restrict__ q, _Float16* __restrict__ k,
                 const int* __restrict__ pos) {
  int t = blockIdx.x * 256 + threadIdx.x;      // [0, S*NH*64)
  int s = t / (NH_ * 64);
  int r = t % (NH_ * 64);
  int h = r >> 6, j = r & 63;
  float p = (float)pos[s];
  float ang = p * __expf(-(float)j * 0.14391156831212787f);  // ln(1e4)/64
  float c = __cosf(ang), sn = __sinf(ang);
  size_t i1 = (size_t)s * H_ + h * 128 + j, i2 = i1 + 64;
  float a = (float)q[i1], b = (float)q[i2];
  q[i1] = (_Float16)(a * c - b * sn); q[i2] = (_Float16)(b * c + a * sn);
  a = (float)k[i1]; b = (float)k[i2];
  k[i1] = (_Float16)(a * c - b * sn); k[i2] = (_Float16)(b * c + a * sn);
}

// Per-row top-k threshold (binary search) + masked softmax.
// Reads fp32 scores, writes f16 probabilities. One 256-thread block per row.
__global__ __launch_bounds__(256)
void topk_softmax(const float* __restrict__ sc, _Float16* __restrict__ pr) {
  __shared__ float red[256];
  __shared__ int  ired[256];
  const int tid = threadIdx.x;
  const float* rp = sc + (size_t)blockIdx.x * S_;
  _Float16* wp = pr + (size_t)blockIdx.x * S_;
  float v[8];
  float lmax = -3.4e38f, lmin = 3.4e38f;
#pragma unroll
  for (int i = 0; i < 8; ++i) {
    v[i] = rp[tid + 256 * i];
    lmax = fmaxf(lmax, v[i]); lmin = fminf(lmin, v[i]);
  }
  red[tid] = lmax; __syncthreads();
  for (int s = 128; s > 0; s >>= 1) { if (tid < s) red[tid] = fmaxf(red[tid], red[tid + s]); __syncthreads(); }
  float hi = red[0]; __syncthreads();
  red[tid] = lmin; __syncthreads();
  for (int s = 128; s > 0; s >>= 1) { if (tid < s) red[tid] = fminf(red[tid], red[tid + s]); __syncthreads(); }
  float lo = red[0]; __syncthreads();

  for (int it = 0; it < 24; ++it) {
    float mid = 0.5f * (lo + hi);
    int c = 0;
#pragma unroll
    for (int i = 0; i < 8; ++i) c += (v[i] >= mid) ? 1 : 0;
    ired[tid] = c; __syncthreads();
    for (int s = 128; s > 0; s >>= 1) { if (tid < s) ired[tid] += ired[tid + s]; __syncthreads(); }
    int cnt = ired[0]; __syncthreads();
    if (cnt >= KK_) lo = mid; else hi = mid;
  }
  float thr = lo;

  float m = 0.0f;  // zeros are present in the sparsified row, so max >= 0
#pragma unroll
  for (int i = 0; i < 8; ++i) { v[i] = (v[i] >= thr) ? v[i] : 0.0f; m = fmaxf(m, v[i]); }
  red[tid] = m; __syncthreads();
  for (int s = 128; s > 0; s >>= 1) { if (tid < s) red[tid] = fmaxf(red[tid], red[tid + s]); __syncthreads(); }
  m = red[0]; __syncthreads();
  float sum = 0.0f;
#pragma unroll
  for (int i = 0; i < 8; ++i) { v[i] = __expf(v[i] - m); sum += v[i]; }
  red[tid] = sum; __syncthreads();
  for (int s = 128; s > 0; s >>= 1) { if (tid < s) red[tid] += red[tid + s]; __syncthreads(); }
  float inv = 1.0f / red[0];
#pragma unroll
  for (int i = 0; i < 8; ++i) wp[tid + 256 * i] = (_Float16)(v[i] * inv);
}

extern "C" void kernel_launch(void* const* d_in, const int* in_sizes, int n_in,
                              void* d_out, int out_size, void* d_ws, size_t ws_size,
                              hipStream_t stream) {
  const float* hs  = (const float*)d_in[0];
  const int*   pos = (const int*)  d_in[1];
  const float* Wq  = (const float*)d_in[2];
  const float* Wk  = (const float*)d_in[3];
  const float* Wv  = (const float*)d_in[4];
  const float* Wo  = (const float*)d_in[5];
  const float* pw  = (const float*)d_in[6];
  float* out = (float*)d_out;

  const size_t SH = (size_t)S_ * H_;
  float*    Sc    = (float*)d_ws;              // [NH,S,S] fp32 scores
  _Float16* hs16  = (_Float16*)(Sc + (size_t)NH_ * S_ * S_);
  _Float16* Wq16  = hs16 + SH;                 // [H,H] each
  _Float16* Wk16  = Wq16 + SH;
  _Float16* Wv16  = Wk16 + SH;
  _Float16* Wo16  = Wv16 + SH;
  _Float16* Q16   = Wo16 + SH;                 // [S,H]
  _Float16* K16   = Q16  + SH;                 // [S,H]
  _Float16* VT16  = K16  + SH;                 // [H,S]  (V transposed)
  _Float16* tWT   = VT16 + SH;                 // [NH,D,D] tanh(pattern)^T
  _Float16* Qp16  = tWT  + (size_t)NH_ * D_ * D_;  // [NH,S,D]
  _Float16* Cx16  = Qp16 + (size_t)NH_ * S_ * D_;  // [S,H] context
  _Float16* P16   = Cx16 + SH;                 // [NH,S,S] f16 probs

  dim3 blk(256);
  const float iscale = 0.08838834764831845f;   // 1/sqrt(128)
  const int NSH = (int)SH;

  // 0) fp32 -> f16 operand conversion (one pass; halves all GEMM traffic)
  f32_to_f16<<<NSH / 1024, blk, 0, stream>>>(hs, hs16, NSH);
  f32_to_f16<<<NSH / 1024, blk, 0, stream>>>(Wq, Wq16, NSH);
  f32_to_f16<<<NSH / 1024, blk, 0, stream>>>(Wk, Wk16, NSH);
  f32_to_f16<<<NSH / 1024, blk, 0, stream>>>(Wv, Wv16, NSH);
  f32_to_f16<<<NSH / 1024, blk, 0, stream>>>(Wo, Wo16, NSH);
  tanhT_kernel<<<(NH_ * D_ * D_) / 256, blk, 0, stream>>>(pw, tWT);

  // 1) Projections: Q,K [S,H];  V stored transposed as VT [H,S]
  dim3 gp(H_ / BN, S_ / BM, 1);
  wmma_gemm<<<gp, blk, 0, stream>>>(hs16, Wq16, Q16,  H_, H_, H_, H_, 0, 0, 0, 1.0f, 0, 0);
  wmma_gemm<<<gp, blk, 0, stream>>>(hs16, Wk16, K16,  H_, H_, H_, H_, 0, 0, 0, 1.0f, 0, 0);
  wmma_gemm<<<gp, blk, 0, stream>>>(hs16, Wv16, VT16, H_, H_, H_, S_, 0, 0, 0, 1.0f, 0, 1);

  // 2) RoPE (in-place on Q16, K16)
  rope_kernel<<<(S_ * NH_ * 64) / 256, blk, 0, stream>>>(Q16, K16, pos);

  // 3) Pattern transform: Qp[h] = Q_h @ tanhW[h]  (B pre-transposed -> A*B^T)
  dim3 gpat(1, S_ / BM, NH_);
  wmma_gemm<<<gpat, blk, 0, stream>>>(Q16, tWT, Qp16, D_,
                                      H_, D_, D_, 128, (long)D_ * D_, (long)S_ * D_, 1.0f, 0, 0);

  // 4) Scores (fp32): Sc[h] = (Qp[h] @ K_h^T) / sqrt(D)
  dim3 gs(S_ / BN, S_ / BM, NH_);
  wmma_gemm<<<gs, blk, 0, stream>>>(Qp16, K16, Sc, D_,
                                    D_, H_, S_, (long)S_ * D_, 128, (long)S_ * S_, iscale, 1, 0);

  // 5) Top-k sparsify + softmax; emit f16 probabilities
  topk_softmax<<<NH_ * S_, blk, 0, stream>>>(Sc, P16);

  // 6) ctx[:, h*D:(h+1)*D] = P[h] @ V_h  ==  P[h] * VT_h^T
  dim3 gav(1, S_ / BM, NH_);
  wmma_gemm<<<gav, blk, 0, stream>>>(P16, VT16, Cx16, S_,
                                     S_, S_, H_, (long)S_ * S_, (long)128 * S_, 128, 1.0f, 0, 0);

  // 7) out (fp32) = ctx @ Wo^T
  wmma_gemm<<<gp, blk, 0, stream>>>(Cx16, Wo16, out, H_, H_, H_, H_, 0, 0, 0, 1.0f, 1, 0);
}
